// MultiHeadAttention_26285199851794
// MI455X (gfx1250) — compile-verified, matrix operand order checked by reading, not run on hardware
//
#include <hip/hip_runtime.h>

// ---------------- problem constants ----------------
#define B_    8
#define M_    1026
#define C_    384
#define H_    6
#define D_    64
#define N1_   513
#define NP_   544         // padded sequence length (multiple of 32)
#define RT_   33          // ceil(513/16) row tiles (16-row granularity)
#define CT_   24          // 384/16 col tiles
#define KT2_  17          // ceil(513/32) key tiles (32-key granularity)

// ---------------- workspace layout (bytes) ----------------
#define OFF_SEL    0
#define OFF_WBF    256
#define WMAT_ELEMS (C_*C_)                          // 147456 per matrix, 7 matrices (TRANSPOSED)
#define OFF_XBF    (OFF_WBF + 7*WMAT_ELEMS*2)       // 2064640
#define XBF_ELEMS  (B_*M_*C_)                       // 3151872
#define OFF_QKV    (OFF_XBF + XBF_ELEMS*2)          // 8368384 (16B aligned)
#define QKVP_ELEMS ((long)B_*NP_*C_)                // 1671168 per tensor; Q,K row-major / V1T,V2T transposed
#define OFF_O      (OFF_QKV + 4*QKVP_ELEMS*2)       // O is B*M*C bf16

// ---------------- WMMA types ----------------
typedef __attribute__((ext_vector_type(16))) __bf16    v16bf;
typedef __attribute__((ext_vector_type(8)))  float     v8f;
typedef __attribute__((ext_vector_type(4)))  unsigned  v4u;

union BfPack { unsigned u[8]; v4u q[2]; v16bf v; };

__device__ __forceinline__ v8f wmma_bf16(const BfPack& a, const BfPack& b, v8f c) {
  return __builtin_amdgcn_wmma_f32_16x16x32_bf16(false, a.v, false, b.v, (short)0, c,
                                                 false, false);
}

__device__ __forceinline__ unsigned short f2bf(float f) {
  unsigned u = __builtin_bit_cast(unsigned, f);
  u += 0x7fffu + ((u >> 16) & 1u);          // round-to-nearest-even
  return (unsigned short)(u >> 16);
}

__device__ __forceinline__ v8f vzero8() {
  v8f z = {0.f,0.f,0.f,0.f,0.f,0.f,0.f,0.f};
  return z;
}

// ================= kernel 0: select + convert (weights transposed) + zero QKV =================
__global__ __launch_bounds__(256) void prep_kernel(
    const float* __restrict__ x,
    const float* __restrict__ w0, const float* __restrict__ w1,
    const float* __restrict__ w2, const float* __restrict__ w3,
    const float* __restrict__ w4, const float* __restrict__ w5,
    const float* __restrict__ w6,
    const float* __restrict__ wts, const float* __restrict__ gmb,
    unsigned char* __restrict__ ws) {
  if (blockIdx.x == 0 && threadIdx.x == 0) {
    float best = wts[0] + gmb[0]; int bi = 0;
    #pragma unroll
    for (int i = 1; i < 4; ++i) {
      float v = wts[i] + gmb[i];
      if (v > best) { best = v; bi = i; }
    }
    *(int*)(ws + OFF_SEL) = bi;             // argmax(softmax(z)) == argmax(z)
  }
  const long tid0  = (long)blockIdx.x * blockDim.x + threadIdx.x;
  const long nthr  = (long)gridDim.x * blockDim.x;

  // zero-fill padded QKV region with 128-bit stores (padding safety for WMMA)
  v4u* zp = (v4u*)(ws + OFF_QKV);
  const long NZ = (4 * QKVP_ELEMS * 2) / 16;
  v4u zv = {0u, 0u, 0u, 0u};
  for (long i = tid0; i < NZ; i += nthr) zp[i] = zv;

  // weights: store TRANSPOSED bf16  Wt[c*384 + k] = W[k*384 + c]
  unsigned short* wbf = (unsigned short*)(ws + OFF_WBF);
  unsigned short* xbf = (unsigned short*)(ws + OFF_XBF);
  const float* Ws[7] = {w0, w1, w2, w3, w4, w5, w6};
  const long NW = 7L * WMAT_ELEMS;
  const long NX = (long)XBF_ELEMS;
  for (long i = tid0; i < NW + NX; i += nthr) {
    if (i < NW) {
      const int  mi = (int)(i / WMAT_ELEMS);
      const long e  = i % WMAT_ELEMS;
      const long c  = e / C_;
      const long k  = e % C_;
      wbf[i] = f2bf(Ws[mi][k * C_ + c]);
    } else {
      xbf[i - NW] = f2bf(x[i - NW]);
    }
  }
}

// ================= kernel 1: Q/K/V1/V2 projections =================
// one wave = one 16x16 output tile; o: 0=Q(sel) 1=K(sel) 2=V1 3=V2
// Q,K stored row-major [b][NP_][C_]; V stored transposed [b][C_][NP_]
__global__ __launch_bounds__(256) void proj_kernel(
    unsigned char* __restrict__ ws,
    const float* __restrict__ bq1, const float* __restrict__ bq2,
    const float* __restrict__ bk1, const float* __restrict__ bk2,
    const float* __restrict__ bv1, const float* __restrict__ bv2) {
  const int wid  = blockIdx.x * 8 + (threadIdx.x >> 5);
  const int lane = threadIdx.x & 31;
  int t = wid;
  const int ct = t % CT_; t /= CT_;
  const int rt = t % RT_; t /= RT_;
  const int b  = t % B_;  t /= B_;
  const int o  = t;                         // 0..3

  const int sel = *(const int*)(ws + OFF_SEL);
  const unsigned short* wbf = (const unsigned short*)(ws + OFF_WBF);
  const unsigned short* xbf = (const unsigned short*)(ws + OFF_XBF);
  unsigned short* qkv = (unsigned short*)(ws + OFF_QKV);

  int widx; long xoff; const float* bias;
  if (o == 0)      { widx = (sel >= 2) ? 1 : 0; xoff = (sel >= 2) ? (long)N1_ * C_ : 0; bias = (sel >= 2) ? bq2 : bq1; }
  else if (o == 1) { widx = 2 + (sel & 1);      xoff = (sel & 1) ? (long)N1_ * C_ : 0;  bias = (sel & 1) ? bk2 : bk1; }
  else if (o == 2) { widx = 4;                  xoff = 0;                               bias = bv1; }
  else             { widx = 5;                  xoff = (long)N1_ * C_;                  bias = bv2; }

  const unsigned short* A  = xbf + (long)b * M_ * C_ + xoff;   // 513 x 384 row-major
  const unsigned short* Wt = wbf + (long)widx * WMAT_ELEMS;    // 384 x 384, transposed (col-major pairs)
  unsigned short* Obuf = qkv + (long)o * QKVP_ELEMS;

  const int hi = lane >> 4, ln = lane & 15, kb = hi * 8;
  const int row = min(rt * 16 + ln, N1_ - 1);    // clamp: pad rows duplicate last row, never stored
  const int col = ct * 16 + ln;

  v8f acc = vzero8();
  const unsigned short* ar = A  + (long)row * C_ + kb;
  const unsigned short* wr = Wt + (long)col * C_ + kb;
  for (int k0 = 0; k0 < C_; k0 += 32) {
    BfPack af, bf;
    af.q[0] = *(const v4u*)(ar + k0);
    af.q[1] = *(const v4u*)(ar + k0 + 16);
    bf.q[0] = *(const v4u*)(wr + k0);
    bf.q[1] = *(const v4u*)(wr + k0 + 16);
    acc = wmma_bf16(af, bf, acc);
  }

  const float bc = bias[col];
  if (o < 2) {                                   // Q,K row-major padded
    unsigned short* dst = Obuf + (long)b * NP_ * C_;
    #pragma unroll
    for (int r = 0; r < 8; ++r) {
      const int grow = rt * 16 + r + kb;
      if (grow < N1_) dst[(long)grow * C_ + col] = f2bf(acc[r] + bc);
    }
  } else {                                       // V transposed [c][key]
    unsigned short* dst = Obuf + (long)b * C_ * NP_;
    #pragma unroll
    for (int r = 0; r < 8; ++r) {
      const int grow = rt * 16 + r + kb;
      if (grow < N1_) dst[(long)col * NP_ + grow] = f2bf(acc[r] + bc);
    }
  }
}

// ================= kernel 2: flash attention (32 keys / iter), A shared by V1,V2 =================
__global__ __launch_bounds__(256) void attn_kernel(unsigned char* __restrict__ ws) {
  __shared__ __align__(16) unsigned short plds[8][512];   // per-wave 16x32 bf16 P tile
  const int wid  = blockIdx.x * 8 + (threadIdx.x >> 5);
  const int lw   = threadIdx.x >> 5;
  const int lane = threadIdx.x & 31;
  int t = wid;
  const int qt = t % RT_; t /= RT_;
  const int h  = t % H_;  t /= H_;
  const int b  = t;

  const unsigned short* qkv = (const unsigned short*)(ws + OFF_QKV);
  const unsigned short* Qb  = qkv + (long)b * NP_ * C_;
  const unsigned short* Kb  = qkv + QKVP_ELEMS     + (long)b * NP_ * C_;
  const unsigned short* V1t = qkv + 2 * QKVP_ELEMS + (long)b * C_ * NP_;
  const unsigned short* V2t = qkv + 3 * QKVP_ELEMS + (long)b * C_ * NP_;
  unsigned short* Ob = (unsigned short*)(ws + OFF_O);

  const int hi = lane >> 4, ln = lane & 15, kb = hi * 8;

  // Q fragments: A layout, rows = queries (zero-padded buffer => unconditional loads)
  const int qrow = qt * 16 + ln;
  BfPack qf[2];
  {
    const unsigned short* qp = Qb + (long)qrow * C_ + h * D_ + kb;
    #pragma unroll
    for (int s = 0; s < 2; ++s) {
      qf[s].q[0] = *(const v4u*)(qp + s * 32);
      qf[s].q[1] = *(const v4u*)(qp + s * 32 + 16);
    }
  }

  float mi[8], li[8];
  v8f o1[4], o2[4];
  #pragma unroll
  for (int r = 0; r < 8; ++r) { mi[r] = -1e30f; li[r] = 0.f; }
  #pragma unroll
  for (int dt = 0; dt < 4; ++dt) { o1[dt] = vzero8(); o2[dt] = vzero8(); }

  for (int kt = 0; kt < KT2_; ++kt) {
    const int ka  = kt * 32 + ln;        // key of S-tile-a column
    const int ka2 = ka + 16;             // key of S-tile-b column
    if (kt + 1 < KT2_)
      __builtin_prefetch((const void*)(Kb + (long)(kt * 32 + 32 + ln) * C_ + h * D_), 0, 0);

    // --- S = Q K^T over 32 keys (two 16-key tiles) ---
    v8f sa = vzero8(), sb = vzero8();
    {
      const unsigned short* kpa = Kb + (long)ka  * C_ + h * D_ + kb;
      const unsigned short* kpb = Kb + (long)ka2 * C_ + h * D_ + kb;
      #pragma unroll
      for (int st = 0; st < 2; ++st) {
        BfPack kfa, kfb;
        kfa.q[0] = *(const v4u*)(kpa + st * 32);
        kfa.q[1] = *(const v4u*)(kpa + st * 32 + 16);
        kfb.q[0] = *(const v4u*)(kpb + st * 32);
        kfb.q[1] = *(const v4u*)(kpb + st * 32 + 16);
        sa = wmma_bf16(qf[st], kfa, sa);
        sb = wmma_bf16(qf[st], kfb, sb);
      }
    }

    // --- online softmax over this 32-key slab ---
    const bool va = ka  < N1_;
    const bool vb = ka2 < N1_;
    float pa8[8], pb8[8];
    #pragma unroll
    for (int r = 0; r < 8; ++r) {
      const float sva = va ? sa[r] * 0.125f : -1e30f;   // scale = d^-0.5
      const float svb = vb ? sb[r] * 0.125f : -1e30f;
      float rmax = fmaxf(sva, svb);
      #pragma unroll
      for (int m = 1; m < 16; m <<= 1) rmax = fmaxf(rmax, __shfl_xor(rmax, m, 32));
      const float mnew  = fmaxf(mi[r], rmax);
      const float alpha = __expf(mi[r] - mnew);
      const float pa = __expf(sva - mnew);
      const float pb = __expf(svb - mnew);
      float rs = pa + pb;
      #pragma unroll
      for (int m = 1; m < 16; m <<= 1) rs += __shfl_xor(rs, m, 32);
      li[r] = li[r] * alpha + rs;
      mi[r] = mnew;
      pa8[r] = pa; pb8[r] = pb;
      #pragma unroll
      for (int dt = 0; dt < 4; ++dt) { o1[dt][r] *= alpha; o2[dt][r] *= alpha; }
    }

    // --- C-layout -> A-layout transpose of P (16x32) via per-wave LDS ---
    #pragma unroll
    for (int r = 0; r < 8; ++r) {
      plds[lw][(r + kb) * 32 + ln]      = f2bf(pa8[r]);
      plds[lw][(r + kb) * 32 + 16 + ln] = f2bf(pb8[r]);
    }
    BfPack pf;
    pf.q[0] = *(const v4u*)(&plds[lw][ln * 32 + kb]);
    pf.q[1] = *(const v4u*)(&plds[lw][ln * 32 + kb + 16]);

    // --- O += P @ V for both value streams (V transposed => b128 loads) ---
    #pragma unroll
    for (int dt = 0; dt < 4; ++dt) {
      const long cbase = (long)(h * D_ + dt * 16 + ln) * NP_ + kt * 32 + kb;
      BfPack vf1, vf2;
      vf1.q[0] = *(const v4u*)(V1t + cbase);
      vf1.q[1] = *(const v4u*)(V1t + cbase + 16);
      vf2.q[0] = *(const v4u*)(V2t + cbase);
      vf2.q[1] = *(const v4u*)(V2t + cbase + 16);
      o1[dt] = wmma_bf16(pf, vf1, o1[dt]);
      o2[dt] = wmma_bf16(pf, vf2, o2[dt]);
    }
  }

  // normalize and scatter: rows [0,N1) get O1, rows [N1,M) get O2
  #pragma unroll
  for (int r = 0; r < 8; ++r) {
    const int q = qt * 16 + r + kb;
    if (q < N1_) {
      const float inv = 1.0f / li[r];
      #pragma unroll
      for (int dt = 0; dt < 4; ++dt) {
        const long c = h * D_ + dt * 16 + ln;
        Ob[((long)b * M_ + q) * C_ + c]       = f2bf(o1[dt][r] * inv);
        Ob[((long)b * M_ + N1_ + q) * C_ + c] = f2bf(o2[dt][r] * inv);
      }
    }
  }
}

// ================= kernel 3: output projection out = O @ Wo + bo (fp32 out) =================
__global__ __launch_bounds__(256) void outproj_kernel(
    const unsigned char* __restrict__ ws, const float* __restrict__ bo,
    float* __restrict__ out) {
  const int wid  = blockIdx.x * 8 + (threadIdx.x >> 5);
  const int lane = threadIdx.x & 31;
  const int ct = wid % CT_;
  const int rt = wid / CT_;                    // 0..512 ; 8208 rows = B*M exactly
  const unsigned short* A  = (const unsigned short*)(ws + OFF_O);
  const unsigned short* Wt = (const unsigned short*)(ws + OFF_WBF) + 6L * WMAT_ELEMS;

  const int hi = lane >> 4, ln = lane & 15, kb = hi * 8;
  const int row = rt * 16 + ln;
  const int col = ct * 16 + ln;

  v8f acc = vzero8();
  const unsigned short* ar = A  + (long)row * C_ + kb;
  const unsigned short* wr = Wt + (long)col * C_ + kb;
  for (int k0 = 0; k0 < C_; k0 += 32) {
    BfPack af, bf;
    af.q[0] = *(const v4u*)(ar + k0);
    af.q[1] = *(const v4u*)(ar + k0 + 16);
    bf.q[0] = *(const v4u*)(wr + k0);
    bf.q[1] = *(const v4u*)(wr + k0 + 16);
    acc = wmma_bf16(af, bf, acc);
  }

  const float bias = bo[col];
  #pragma unroll
  for (int r = 0; r < 8; ++r)
    out[((long)rt * 16 + r + kb) * C_ + col] = acc[r] + bias;
}

// ================= host launcher =================
extern "C" void kernel_launch(void* const* d_in, const int* in_sizes, int n_in,
                              void* d_out, int out_size, void* d_ws, size_t ws_size,
                              hipStream_t stream) {
  const float* x   = (const float*)d_in[0];
  const float* Wq1 = (const float*)d_in[1];  const float* bq1 = (const float*)d_in[2];
  const float* Wq2 = (const float*)d_in[3];  const float* bq2 = (const float*)d_in[4];
  const float* Wk1 = (const float*)d_in[5];  const float* bk1 = (const float*)d_in[6];
  const float* Wk2 = (const float*)d_in[7];  const float* bk2 = (const float*)d_in[8];
  const float* Wv1 = (const float*)d_in[9];  const float* bv1 = (const float*)d_in[10];
  const float* Wv2 = (const float*)d_in[11]; const float* bv2 = (const float*)d_in[12];
  const float* Wo  = (const float*)d_in[13]; const float* bo  = (const float*)d_in[14];
  const float* wts = (const float*)d_in[15];
  const float* gmb = (const float*)d_in[16];
  float* out = (float*)d_out;
  unsigned char* ws = (unsigned char*)d_ws;

  prep_kernel<<<2048, 256, 0, stream>>>(x, Wq1, Wq2, Wk1, Wk2, Wv1, Wv2, Wo, wts, gmb, ws);

  // 4 outputs * 8 batches * 33 row tiles * 24 col tiles = 25344 waves / 8 per block
  proj_kernel<<<3168, 256, 0, stream>>>(ws, bq1, bq2, bk1, bk2, bv1, bv2);

  // 8 batches * 6 heads * 33 query tiles = 1584 waves / 8 per block
  attn_kernel<<<198, 256, 0, stream>>>(ws);

  // 513 row tiles * 24 col tiles = 12312 waves / 8 per block
  outproj_kernel<<<1539, 256, 0, stream>>>(ws, bo, out);
}
